// SFUnit_72000831750803
// MI455X (gfx1250) — matrix operand order repaired and checked by reading, not compiled, not run on hardware
//
#include <hip/hip_runtime.h>
#include <math.h>

#define TOTAL 66832

typedef __attribute__((ext_vector_type(16))) __bf16 v16bf;
typedef __attribute__((ext_vector_type(8)))  float  v8f;
typedef __attribute__((ext_vector_type(4)))  unsigned int v4u;
typedef __attribute__((ext_vector_type(8)))  unsigned int v8u;
typedef unsigned short u16;

union AB16 { v4u q[2]; v8u o; v16bf bf; };

__device__ __forceinline__ u16 f2bf(float f){
  unsigned int u = __float_as_uint(f);
  unsigned int r = u + 0x7FFFu + ((u >> 16) & 1u);
  return (u16)(r >> 16);
}
__device__ __forceinline__ float gelu_f(float x){
  return 0.5f * x * (1.0f + erff(x * 0.70710678118f));
}
// A fragment: row-major A[M][K], lda multiple of 8. Lane<16: K k0+0..7 & k0+16..23;
// lane>=16: K k0+8..15 & k0+24..31 (ISA 7.12.2 16-bit A layout).
__device__ __forceinline__ v16bf load_a_frag(const u16* A, int lda, int row, int k0, int lane){
  const int hi = (lane >> 4) << 3;
  const u16* p0 = A + row*lda + k0 + hi;
  AB16 t; t.q[0] = *(const v4u*)p0; t.q[1] = *(const v4u*)(p0 + 16);
  return t.bf;
}
// B fragment from pre-transposed Bt[N][K], ldb multiple of 16.
// Lane<16: K k0+0..7 & k0+16..23 ; lane>=16: K k0+8..15 & k0+24..31, column = lane&15.
__device__ __forceinline__ v16bf load_b_frag(const u16* Bt, int ldb, int col, int k0, int lane){
  const int hi = (lane >> 4) << 3;
  const u16* p0 = Bt + col*ldb + k0 + hi;
  AB16 t; t.q[0] = *(const v4u*)p0; t.q[1] = *(const v4u*)(p0 + 16);
  return t.bf;
}
__device__ __forceinline__ v8f wmma_bf16(v16bf a, v16bf b, v8f c){
  return __builtin_amdgcn_wmma_f32_16x16x32_bf16(false, a, false, b, (short)0, c, false, false);
}
__device__ __forceinline__ float dot64(const float4* __restrict__ w, const float* q){
  float s = 0.f;
  #pragma unroll
  for(int i = 0; i < 16; i++){
    float4 v = w[i];
    s += v.x*q[4*i] + v.y*q[4*i+1] + v.z*q[4*i+2] + v.w*q[4*i+3];
  }
  return s;
}

// ---------------- layout / transpose kernels ----------------
__global__ void k_w2T(const float* __restrict__ src, float* __restrict__ dst){
  int e = blockIdx.x*256 + threadIdx.x;
  if (e >= 64*TOTAL) return;
  int j = e / TOTAL, idx = e - j*TOTAL;
  dst[idx*64 + j] = src[e];
}
__global__ void k_wT(const float* __restrict__ src, u16* __restrict__ dst, int K, int N){
  int e = blockIdx.x*256 + threadIdx.x;
  if (e >= K*N) return;
  int k = e / N, n = e - k*N;
  dst[n*K + k] = f2bf(src[e]);
}
__global__ void k_imgT(const float* __restrict__ img, float* __restrict__ imgT){
  int bidx = blockIdx.x;            // b*256*64 + c*64 + y
  int y = bidx & 63; int bc = bidx >> 6; int c = bc & 255; int b = bc >> 8;
  int x = threadIdx.x;              // 64 threads
  imgT[(((b*64 + y)*64) + x)*256 + c] = img[(((b*256 + c)*64) + y)*64 + x];
}

// ---------------- prep: LN + roi adjust + sampling points + q1 ----------------
__global__ void __launch_bounds__(256) k_prep(
    const float* __restrict__ embed, const float* __restrict__ roi,
    const float* __restrict__ g_adj, const float* __restrict__ b_adj, const float* __restrict__ w_adj,
    const float* __restrict__ g_off, const float* __restrict__ b_off, const float* __restrict__ w_off,
    const float* __restrict__ off_bias,
    const float* __restrict__ g_pg, const float* __restrict__ b_pg,
    const float* __restrict__ pg_w1, const float* __restrict__ pg_b1,
    float* __restrict__ q1g, float* __restrict__ ptsg)
{
  __shared__ float red[256];
  __shared__ float xs[256];
  __shared__ float offv[72];
  __shared__ float logitv[4];
  __shared__ float nroi[4];
  __shared__ float musd[4];
  __shared__ float stats[2];
  int t = blockIdx.x, tid = threadIdx.x;
  float v = embed[t*256 + tid];
  red[tid] = v; __syncthreads();
  for(int s = 128; s > 0; s >>= 1){ if (tid < s) red[tid] += red[tid+s]; __syncthreads(); }
  if (tid == 0) stats[0] = red[0] * (1.f/256.f);
  __syncthreads();
  float m = stats[0], dv = v - m;
  red[tid] = dv*dv; __syncthreads();
  for(int s = 128; s > 0; s >>= 1){ if (tid < s) red[tid] += red[tid+s]; __syncthreads(); }
  if (tid == 0) stats[1] = rsqrtf(red[0] * (1.f/256.f) + 1e-5f);
  __syncthreads();
  xs[tid] = dv * stats[1];
  __syncthreads();
  if (tid < 4){
    float acc = 0.f;
    for(int d = 0; d < 256; d++) acc += (xs[d]*g_adj[d] + b_adj[d]) * w_adj[d*4 + tid];
    logitv[tid] = acc;
  } else if (tid < 76){
    int k = tid - 4; float acc = off_bias[k];
    for(int d = 0; d < 256; d++) acc += (xs[d]*g_off[d] + b_off[d]) * w_off[d*72 + k];
    offv[k] = acc;
  } else if (tid < 140){
    int qj = tid - 76; float acc = pg_b1[qj];
    for(int d = 0; d < 256; d++) acc += (xs[d]*g_pg[d] + b_pg[d]) * pg_w1[d*64 + qj];
    q1g[t*64 + qj] = acc;
  }
  __syncthreads();
  if (tid < 2){
    int i = tid;
    float r0 = roi[t*4 + i], r2 = roi[t*4 + 2 + i];
    float cx = (r2 + r0)*0.5f, cw = fabsf(r2 - r0);
    cx += logitv[i]*cw;
    cw *= __expf(logitv[2 + i]);
    nroi[i] = cx - 0.5f*cw;
    nroi[2 + i] = cx + 0.5f*cw;
    float mu = 0.f;
    for(int p = 0; p < 36; p++) mu += offv[p*2 + i];
    mu *= (1.f/36.f);
    float sd = 0.f;
    for(int p = 0; p < 36; p++){ float dd = offv[p*2 + i] - mu; sd += dd*dd; }
    sd = sqrtf(sd * (1.f/35.f)) + 1e-7f;
    musd[i] = mu; musd[2 + i] = sd;
  }
  __syncthreads();
  if (tid < 72){
    int i = tid & 1;
    float offn = (offv[tid] - musd[i]) / (3.f * musd[2 + i]);
    float rxy = (nroi[2 + i] + nroi[i]) * 0.5f;
    float rwh = nroi[2 + i] - nroi[i];
    ptsg[t*72 + tid] = rxy + offn * rwh;
  }
}

// ---------------- bilinear grid sample -> bf16 A matrix [t][48][256] ----------------
__global__ void __launch_bounds__(256) k_sample(const float* __restrict__ imgT,
                                                const float* __restrict__ pts,
                                                u16* __restrict__ sampledA){
  int bp = blockIdx.x;            // t*48 + p
  int t = bp / 48, p = bp - t*48;
  int c = threadIdx.x;
  u16* dst = sampledA + (size_t)bp * 256;
  if (p >= 36){ dst[c] = 0; return; }
  int b = t >> 6;
  float px = pts[t*72 + p*2], py = pts[t*72 + p*2 + 1];
  float x = px*64.f - 0.5f, y = py*64.f - 0.5f;
  float x0 = floorf(x), y0 = floorf(y);
  float wx = x - x0, wy = y - y0;
  int x0i = (int)x0, y0i = (int)y0;
  const float* base = imgT + (size_t)b*64*64*256;
  float acc = 0.f;
  #pragma unroll
  for(int q = 0; q < 4; q++){
    int xi = x0i + (q & 1), yi = y0i + (q >> 1);
    float w = ((q & 1) ? wx : 1.f - wx) * ((q >> 1) ? wy : 1.f - wy);
    bool valid = (xi >= 0) && (xi < 64) && (yi >= 0) && (yi < 64);
    int xc = xi < 0 ? 0 : (xi > 63 ? 63 : xi);
    int yc = yi < 0 ? 0 : (yi > 63 ? 63 : yi);
    float vv = base[(yc*64 + xc)*256 + c];
    acc += vv * (valid ? w : 0.f);
  }
  dst[c] = f2bf(acc);
}

// ---------------- fused per-token adaptive mixing (WMMA) ----------------
__global__ void __launch_bounds__(256) k_mix_token(
    const u16* __restrict__ sampledA,     // [1024][48][256] bf16
    const float* __restrict__ w2T,        // [TOTAL][64]
    const float* __restrict__ pg_b2,      // [TOTAL]
    const float* __restrict__ q1g,        // [1024][64]
    const float* __restrict__ m_beta,     // [256]
    const float* __restrict__ s_beta,     // [36]
    u16* __restrict__ mixA)               // [1024][9216] bf16
{
  __shared__ float q1s[64];
  __shared__ __attribute__((aligned(32))) u16 smA[48*64];     // [o][p] zero-padded
  __shared__ __attribute__((aligned(32))) u16 cmT[32*256];    // panel: [d'][c]
  __shared__ __attribute__((aligned(32))) u16 out1T[256*64];  // [d][p] zero-padded
  const int t = blockIdx.x;
  const int tid = threadIdx.x;
  const int lane = tid & 31;
  const int wave = tid >> 5;
  if (tid < 64) q1s[tid] = q1g[t*64 + tid];
  for(int i = tid; i < 256*64; i += 256) out1T[i] = 0;
  __syncthreads();
  // generate sparse-shape A: sm[o][p] (36x36 valid), zero padded to 48x64
  for(int e = tid; e < 48*64; e += 256){
    int o = e >> 6, p = e & 63;
    float v = 0.f;
    if (o < 36 && p < 36){
      int idx = 65536 + o*36 + p;
      v = pg_b2[idx] + dot64((const float4*)(w2T + (size_t)idx*64), q1s);
    }
    smA[e] = f2bf(v);
  }
  const u16* At = sampledA + (size_t)t*48*256;
  // 8 panels of 32 output channels
  for(int panel = 0; panel < 8; panel++){
    __syncthreads();
    for(int e = tid; e < 32*256; e += 256){
      int c = e & 255, dd = e >> 8;
      int idx = c*256 + (panel*32 + dd);
      float v = pg_b2[idx] + dot64((const float4*)(w2T + (size_t)idx*64), q1s);
      cmT[dd*256 + c] = f2bf(v);
    }
    __syncthreads();
    if (wave < 6){
      int mt = wave >> 1, nt = wave & 1;
      int arow = mt*16 + (lane & 15);
      int bcol = nt*16 + (lane & 15);
      v8f acc = {};
      for(int k0 = 0; k0 < 256; k0 += 32){
        v16bf a = load_a_frag(At, 256, arow, k0, lane);
        v16bf b = load_b_frag(cmT, 256, bcol, k0, lane);
        acc = wmma_bf16(a, b, acc);
      }
      int prow0 = mt*16 + ((lane >> 4) << 3);
      int d = panel*32 + nt*16 + (lane & 15);
      float mb = m_beta[d];
      #pragma unroll
      for(int r = 0; r < 8; r++){
        int p = prow0 + r;
        if (p < 36) out1T[d*64 + p] = f2bf(gelu_f(acc[r] + mb));
      }
    }
  }
  __syncthreads();
  // out2 = sm @ out1 : M=48(pad of 36 o), K=64(pad of 36 p), N=256
  for(int tw = wave; tw < 48; tw += 8){
    int mt = tw >> 4, nt = tw & 15;
    int arow = mt*16 + (lane & 15);
    int bcol = nt*16 + (lane & 15);
    v8f acc = {};
    for(int k0 = 0; k0 < 64; k0 += 32){
      v16bf a = load_a_frag(smA, 64, arow, k0, lane);
      v16bf b = load_b_frag(out1T, 64, bcol, k0, lane);
      acc = wmma_bf16(a, b, acc);
    }
    int orow0 = mt*16 + ((lane >> 4) << 3);
    int d = nt*16 + (lane & 15);
    #pragma unroll
    for(int r = 0; r < 8; r++){
      int o = orow0 + r;
      if (o < 36){
        float v = gelu_f(acc[r] + s_beta[o]);
        mixA[(size_t)t*9216 + o*256 + d] = f2bf(v);
      }
    }
  }
}

// ---------------- generic WMMA GEMM: C[M,N] = A_bf16[M,K] * Bt_bf16[N,K] ----------------
// MODE 0: +bias -> f32 ; MODE 1: +bias+residual -> f32 ; MODE 2: gelu(+bias) -> bf16
template<int MODE>
__global__ void __launch_bounds__(256) k_gemm(const u16* __restrict__ A, const u16* __restrict__ Bt,
                                              const float* __restrict__ bias, const float* __restrict__ resid,
                                              float* __restrict__ Cf, u16* __restrict__ Cbf,
                                              int M, int N, int K){
  const int lane = threadIdx.x & 31;
  const int wave = threadIdx.x >> 5;
  const int mt = wave & 3, nt = wave >> 2;
  const int row0 = blockIdx.x*64 + mt*16;
  const int col0 = blockIdx.y*32 + nt*16;
  const int arow = row0 + (lane & 15);
  const int bcol = col0 + (lane & 15);
  v8f acc = {};
  for(int k0 = 0; k0 < K; k0 += 32){
    v16bf a = load_a_frag(A, K, arow, k0, lane);
    v16bf b = load_b_frag(Bt, K, bcol, k0, lane);
    acc = wmma_bf16(a, b, acc);
  }
  const int rbase = row0 + ((lane >> 4) << 3);
  const int col = col0 + (lane & 15);
  const float bs = bias[col];
  #pragma unroll
  for(int r = 0; r < 8; r++){
    int row = rbase + r;
    float v = acc[r] + bs;
    if (MODE == 1) v += resid[(size_t)row*N + col];
    if (MODE == 2) Cbf[(size_t)row*N + col] = f2bf(gelu_f(v));
    else           Cf [(size_t)row*N + col] = v;
  }
}

// ---------------- LayerNorm -> bf16 ----------------
__global__ void __launch_bounds__(256) k_ln(const float* __restrict__ x, const float* __restrict__ g,
                                            const float* __restrict__ b, u16* __restrict__ out){
  __shared__ float red[256];
  __shared__ float stats[2];
  int t = blockIdx.x, tid = threadIdx.x;
  float v = x[t*256 + tid];
  red[tid] = v; __syncthreads();
  for(int s = 128; s > 0; s >>= 1){ if (tid < s) red[tid] += red[tid+s]; __syncthreads(); }
  if (tid == 0) stats[0] = red[0] * (1.f/256.f);
  __syncthreads();
  float m = stats[0], dv = v - m;
  red[tid] = dv*dv; __syncthreads();
  for(int s = 128; s > 0; s >>= 1){ if (tid < s) red[tid] += red[tid+s]; __syncthreads(); }
  if (tid == 0) stats[1] = rsqrtf(red[0] * (1.f/256.f) + 1e-5f);
  __syncthreads();
  out[t*256 + tid] = f2bf(dv * stats[1] * g[tid] + b[tid]);
}

// ---------------- attention per (batch, head) ----------------
__global__ void __launch_bounds__(256) k_attn(const float* __restrict__ qkv, u16* __restrict__ obuf){
  __shared__ __attribute__((aligned(32))) u16 qA[64*32];
  __shared__ __attribute__((aligned(32))) u16 kA[64*32];
  __shared__ __attribute__((aligned(32))) u16 vT[32*64];
  __shared__ float S[64*64];
  __shared__ __attribute__((aligned(32))) u16 Pb[64*64];
  const int bh = blockIdx.x;
  const int b = bh >> 3, h = bh & 7;
  const int tid = threadIdx.x, lane = tid & 31, wave = tid >> 5;
  for(int e = tid; e < 64*32; e += 256){
    int n = e >> 5, hd = e & 31;
    int src = (b*64 + n)*768 + h*32 + hd;
    qA[n*32 + hd] = f2bf(qkv[src]);
    kA[n*32 + hd] = f2bf(qkv[src + 256]);
    vT[hd*64 + n] = f2bf(qkv[src + 512]);
  }
  __syncthreads();
  #pragma unroll
  for(int i = 0; i < 2; i++){
    int tw = wave*2 + i;
    int mt = tw >> 2, nt = tw & 3;
    int arow = mt*16 + (lane & 15);
    int bcol = nt*16 + (lane & 15);
    v8f acc = {};
    v16bf a = load_a_frag(qA, 32, arow, 0, lane);
    v16bf bb = load_b_frag(kA, 32, bcol, 0, lane);
    acc = wmma_bf16(a, bb, acc);
    int r0 = mt*16 + ((lane >> 4) << 3);
    int col = nt*16 + (lane & 15);
    #pragma unroll
    for(int r = 0; r < 8; r++) S[(r0 + r)*64 + col] = acc[r] * 0.176776695f; // 1/sqrt(32)
  }
  __syncthreads();
  if (tid < 64){
    float mx = -1e30f;
    for(int j = 0; j < 64; j++) mx = fmaxf(mx, S[tid*64 + j]);
    float sum = 0.f;
    for(int j = 0; j < 64; j++){ float e = __expf(S[tid*64 + j] - mx); S[tid*64 + j] = e; sum += e; }
    float inv = 1.f / sum;
    for(int j = 0; j < 64; j++) Pb[tid*64 + j] = f2bf(S[tid*64 + j] * inv);
  }
  __syncthreads();
  {
    int mt = wave >> 1, nt = wave & 1;
    int arow = mt*16 + (lane & 15);
    int bcol = nt*16 + (lane & 15);
    v8f acc = {};
    for(int k0 = 0; k0 < 64; k0 += 32){
      v16bf a = load_a_frag(Pb, 64, arow, k0, lane);
      v16bf bb = load_b_frag(vT, 64, bcol, k0, lane);
      acc = wmma_bf16(a, bb, acc);
    }
    int r0 = mt*16 + ((lane >> 4) << 3);
    int col = h*32 + nt*16 + (lane & 15);
    #pragma unroll
    for(int r = 0; r < 8; r++)
      obuf[(size_t)(b*64 + r0 + r)*256 + col] = f2bf(acc[r]);
  }
}

// ---------------- host launch ----------------
extern "C" void kernel_launch(void* const* d_in, const int* in_sizes, int n_in,
                              void* d_out, int out_size, void* d_ws, size_t ws_size,
                              hipStream_t stream){
  (void)in_sizes; (void)n_in; (void)out_size; (void)ws_size;
  const float* img_feat = (const float*)d_in[0];
  const float* embed    = (const float*)d_in[1];
  const float* roi      = (const float*)d_in[2];
  const float* ln_adj_g = (const float*)d_in[3];
  const float* ln_adj_b = (const float*)d_in[4];
  const float* w_adj    = (const float*)d_in[5];
  const float* ln_off_g = (const float*)d_in[6];
  const float* ln_off_b = (const float*)d_in[7];
  const float* w_off    = (const float*)d_in[8];
  const float* off_bias = (const float*)d_in[9];
  const float* pg_ln_g  = (const float*)d_in[10];
  const float* pg_ln_b  = (const float*)d_in[11];
  const float* pg_w1    = (const float*)d_in[12];
  const float* pg_b1    = (const float*)d_in[13];
  const float* pg_w2    = (const float*)d_in[14];
  const float* pg_b2    = (const float*)d_in[15];
  const float* m_beta   = (const float*)d_in[16];
  const float* s_beta   = (const float*)d_in[17];
  const float* mix_w    = (const float*)d_in[18];
  const float* mix_b    = (const float*)d_in[19];
  const float* ln1_g    = (const float*)d_in[20];
  const float* ln1_b    = (const float*)d_in[21];
  const float* qkv_w    = (const float*)d_in[22];
  const float* qkv_b    = (const float*)d_in[23];
  const float* proj_w   = (const float*)d_in[24];
  const float* proj_b   = (const float*)d_in[25];
  const float* ln2_g    = (const float*)d_in[26];
  const float* ln2_b    = (const float*)d_in[27];
  const float* fc1_w    = (const float*)d_in[28];
  const float* fc1_b    = (const float*)d_in[29];
  const float* fc2_w    = (const float*)d_in[30];
  const float* fc2_b    = (const float*)d_in[31];

  char* ws = (char*)d_ws;
  float* w2T   = (float*)(ws + 0);                 // 17,108,992 B
  u16*  qkvT   = (u16*)(ws + 17108992);            //    393,216
  u16*  projT  = (u16*)(ws + 17502208);            //    131,072
  u16*  fc1T   = (u16*)(ws + 17633280);            //    524,288
  u16*  fc2T   = (u16*)(ws + 18157568);            //    524,288
  u16*  mixT   = (u16*)(ws + 18681856);            //  4,718,592
  float* q1    = (float*)(ws + 23400448);          //    262,144
  float* pts   = (float*)(ws + 23662592);          //    294,912
  float* imgT  = (float*)(ws + 23957504);          // 67,108,864
  u16*  sampled= (u16*)(ws + 91066368);            // 25,165,824 (end 116,232,192)
  // aliases of dead regions:
  u16*  mixA   = (u16*)(ws + 23957504);            // 18.9 MB, over imgT (dead after sampling)
  char* misc   = ws + 91066368;                    // over sampled (dead after k_mix_token)
  float* embed2 = (float*)(misc + 0);
  u16*  xA      = (u16*) (misc + 1048576);
  float* qkvbuf = (float*)(misc + 1572864);
  u16*  obuf    = (u16*) (misc + 4718592);
  float* embed3 = (float*)(misc + 5242880);
  u16*  xA2     = (u16*) (misc + 6291456);
  u16*  hbuf    = (u16*) (misc + 6815744);

  // ---- one-time layout conversions (L2-resident thereafter) ----
  k_w2T<<<(64*TOTAL + 255)/256, 256, 0, stream>>>(pg_w2, w2T);
  k_wT<<<(256*768  + 255)/256, 256, 0, stream>>>(qkv_w, qkvT, 256, 768);
  k_wT<<<(256*256  + 255)/256, 256, 0, stream>>>(proj_w, projT, 256, 256);
  k_wT<<<(256*1024 + 255)/256, 256, 0, stream>>>(fc1_w, fc1T, 256, 1024);
  k_wT<<<(1024*256 + 255)/256, 256, 0, stream>>>(fc2_w, fc2T, 1024, 256);
  k_wT<<<(9216*256 + 255)/256, 256, 0, stream>>>(mix_w, mixT, 9216, 256);
  k_imgT<<<16*256*64, 64, 0, stream>>>(img_feat, imgT);

  // ---- prep + sampling ----
  k_prep<<<1024, 256, 0, stream>>>(embed, roi, ln_adj_g, ln_adj_b, w_adj,
                                   ln_off_g, ln_off_b, w_off, off_bias,
                                   pg_ln_g, pg_ln_b, pg_w1, pg_b1, q1, pts);
  k_sample<<<1024*48, 256, 0, stream>>>(imgT, pts, sampled);

  // ---- fused adaptive mixing ----
  k_mix_token<<<1024, 256, 0, stream>>>(sampled, w2T, pg_b2, q1, m_beta, s_beta, mixA);

  // ---- mix projection + residual ----
  k_gemm<1><<<dim3(16, 8), 256, 0, stream>>>(mixA, mixT, mix_b, embed, embed2, nullptr, 1024, 256, 9216);

  // ---- attention ----
  k_ln<<<1024, 256, 0, stream>>>(embed2, ln1_g, ln1_b, xA);
  k_gemm<0><<<dim3(16, 24), 256, 0, stream>>>(xA, qkvT, qkv_b, nullptr, qkvbuf, nullptr, 1024, 768, 256);
  k_attn<<<128, 256, 0, stream>>>(qkvbuf, obuf);
  k_gemm<1><<<dim3(16, 8), 256, 0, stream>>>(obuf, projT, proj_b, embed2, embed3, nullptr, 1024, 256, 256);

  // ---- MLP ----
  k_ln<<<1024, 256, 0, stream>>>(embed3, ln2_g, ln2_b, xA2);
  k_gemm<2><<<dim3(16, 32), 256, 0, stream>>>(xA2, fc1T, fc1_b, nullptr, nullptr, hbuf, 1024, 1024, 256);
  k_gemm<1><<<dim3(16, 8), 256, 0, stream>>>(hbuf, fc2T, fc2_b, embed3, (float*)d_out, nullptr, 1024, 256, 1024);
}
